// EncoderTrans_10333691314738
// MI455X (gfx1250) — matrix operand order repaired and checked by reading, not compiled, not run on hardware
//
#include <hip/hip_runtime.h>
#include <hip/hip_bf16.h>
#include <math.h>

// ---------------------------------------------------------------------------
// Problem constants (from reference)
// ---------------------------------------------------------------------------
#define BATCH   4
#define SEQ     2048
#define DMODEL  512
#define NHEADS  8
#define DK      64
#define DFF     1024
#define NTOK    (BATCH * SEQ)          // 8192 rows
#define LN_EPS  1e-5f

typedef __attribute__((ext_vector_type(16))) __bf16 v16bf;
typedef __attribute__((ext_vector_type(8)))  __bf16 v8bf;
typedef __attribute__((ext_vector_type(8)))  float  v8f;

// Per-32-column involution: stored order [0..7, 16..23, 8..15, 24..31].
// A tensor stored with perm'd columns yields A-fragments as ONE contiguous
// 32B load per lane (hi=0 -> pos 0..15 = K[0..7,16..23]; hi=1 -> pos 16..31).
__device__ __forceinline__ int permcol(int w)
{
    return (w & 7) | ((w & 16) >> 1) | ((w & 8) << 1);
}

// LDS byte-offset of a generic pointer to a __shared__ object.
typedef const void __attribute__((address_space(3))) lds_cvoid;
__device__ __forceinline__ unsigned lds_off_u32(const void* p)
{
    return (unsigned)(unsigned long long)(lds_cvoid*)p;
}

// CDNA5 async global->LDS copy (per-lane 16B), tracked by ASYNCcnt.
__device__ __forceinline__ void async_load_lds_b128(unsigned ldsoff, const void* gptr)
{
    asm volatile("global_load_async_to_lds_b128 %0, %1, off"
                 :: "v"(ldsoff), "v"(gptr)
                 : "memory");
}

// ---------------------------------------------------------------------------
// Kernel 1: embedding * sqrt(D) + positional encoding; f32 natural order +
// bf16 A-permuted copy, plus key-pad mask (float) into the output tail.
// ---------------------------------------------------------------------------
__global__ __launch_bounds__(256) void
k_prep(const int* __restrict__ tokens,
       const float* __restrict__ emb,
       float* __restrict__ xf,
       __bf16* __restrict__ xbf,
       float* __restrict__ maskout)
{
    const int row = blockIdx.x;            // global token index b*SEQ + t
    const int t   = row & (SEQ - 1);       // position within sequence
    const int tok = tokens[row];
    const float scale = sqrtf((float)DMODEL);

    #pragma unroll
    for (int j = 0; j < 2; ++j) {
        const int d = threadIdx.x * 2 + j;
        const int i2 = d & ~1;
        // 10000^(-i2/512) == exp2(-i2 * log2(10000)/512)
        const float freq = exp2f(-(float)i2 * (13.2877123795f / (float)DMODEL));
        const float ang  = (float)t * freq;
        const float pe   = (d & 1) ? __cosf(ang) : __sinf(ang);
        const float v    = emb[(size_t)tok * DMODEL + d] * scale + pe;
        xf[(size_t)row * DMODEL + d] = v;
        const int dp = (d & ~31) | permcol(d & 31);
        xbf[(size_t)row * DMODEL + dp] = (__bf16)v;
    }
    if (threadIdx.x == 0 && maskout != nullptr) {
        maskout[row] = (tok == 0) ? 1.0f : 0.0f;
    }
}

// ---------------------------------------------------------------------------
// Kernel 2: f32 weight (K x N row-major) -> bf16 B-fragment-swizzled layout:
// flat[((tile_n*(K/32) + k32)*32 + lane)*16 + e] (contiguous 32B per lane).
// ---------------------------------------------------------------------------
template<int N, int K>
__global__ __launch_bounds__(256) void
k_swizzle_w(const float* __restrict__ W, __bf16* __restrict__ out)
{
    const int gid = blockIdx.x * blockDim.x + threadIdx.x;
    constexpr int KT = K / 32;
    constexpr int TOTAL = (N / 16) * KT * 32;
    if (gid >= TOTAL) return;

    const int lane   = gid & 31;
    const int rest   = gid >> 5;
    const int k32    = rest % KT;
    const int tile_n = rest / KT;
    const int n  = lane & 15;
    const int hi = lane >> 4;
    const int col   = tile_n * 16 + n;
    const int kbase = k32 * 32 + hi * 8;

    v8bf t0, t1;
    #pragma unroll
    for (int i = 0; i < 4; ++i) {
        t0[2 * i]     = (__bf16)W[(size_t)(kbase + 2 * i)     * N + col];
        t0[2 * i + 1] = (__bf16)W[(size_t)(kbase + 2 * i + 1) * N + col];
        t1[2 * i]     = (__bf16)W[(size_t)(kbase + 16 + 2 * i)     * N + col];
        t1[2 * i + 1] = (__bf16)W[(size_t)(kbase + 16 + 2 * i + 1) * N + col];
    }
    v8bf* dst = (v8bf*)(out + (size_t)gid * 16);
    dst[0] = t0;
    dst[1] = t1;
}

// ---------------------------------------------------------------------------
// Kernel 3: swizzle V [NTOK, DMODEL] bf16 (natural order) into attention
// B-fragment layout: tile = (((b*(SEQ/32)+k32)*NHEADS + h)*4 + t4),
// flat[tile*512 + lane*16 + e].
// ---------------------------------------------------------------------------
__global__ __launch_bounds__(256) void
k_swizzle_v(const __bf16* __restrict__ V, __bf16* __restrict__ out)
{
    const int gid = blockIdx.x * blockDim.x + threadIdx.x;   // TOTAL = 262144
    const int lane = gid & 31;
    const int t4   = (gid >> 5) & 3;
    const int h    = (gid >> 7) & 7;
    const int rest = gid >> 10;           // b*(SEQ/32) + k32
    const int k32  = rest & 63;
    const int b    = rest >> 6;
    const int n  = lane & 15;
    const int hi = lane >> 4;

    const __bf16* src =
        V + (size_t)(b * SEQ + k32 * 32 + hi * 8) * DMODEL + h * DK + t4 * 16 + n;
    v8bf t0, t1;
    #pragma unroll
    for (int i = 0; i < 4; ++i) {
        t0[2 * i]     = src[(size_t)(2 * i)     * DMODEL];
        t0[2 * i + 1] = src[(size_t)(2 * i + 1) * DMODEL];
        t1[2 * i]     = src[(size_t)(16 + 2 * i)     * DMODEL];
        t1[2 * i + 1] = src[(size_t)(16 + 2 * i + 1) * DMODEL];
    }
    v8bf* dst = (v8bf*)(out + (size_t)gid * 16);
    dst[0] = t0;
    dst[1] = t1;
}

// ---------------------------------------------------------------------------
// Kernel 4: WMMA GEMM, C = A(MxK, bf16, A-permuted rows) * B(swizzled) + bias.
// One wave computes a 32(M) x 64(N) block; 8 wmma per k-step.
// All fragment loads are single contiguous v16bf (no repacking movs).
// PERMBF: write the bf16 output with A-permuted columns (for tensors that
// are consumed as A-fragments downstream).
// ---------------------------------------------------------------------------
template<int N, int K, bool RELU, bool WF32, bool WBF16, bool PERMBF>
__global__ __launch_bounds__(128, 1) void
k_gemm(const __bf16* __restrict__ A,
       const __bf16* __restrict__ Bsw,
       const float* __restrict__ bias,
       float* __restrict__ Cf,
       __bf16* __restrict__ Cbf)
{
    const int lane    = threadIdx.x;                 // 0..31
    const int pair_m  = blockIdx.y * blockDim.y + threadIdx.y;   // 32-row pair
    const int strip_n = blockIdx.x;                  // 64-wide N strip
    const int n    = lane & 15;
    const int hi   = lane >> 4;

    constexpr int KT = K / 32;
    v8f acc[2][4] = {};

    const __bf16* arow0 = A + (size_t)(pair_m * 32 + n) * K + hi * 16;    // m-tile 0
    const __bf16* arow1 = arow0 + (size_t)16 * K;                         // m-tile 1
    const __bf16* bbase = Bsw + ((size_t)(strip_n * 4) * KT) * 32 * 16;

    for (int k32 = 0; k32 < KT; ++k32) {
        // ---- two A fragments: single contiguous 32B loads (perm'd storage)
        const v16bf af0 = *(const v16bf*)(arow0 + k32 * 32);
        const v16bf af1 = *(const v16bf*)(arow1 + k32 * 32);
        if (k32 + 1 < KT) {
            __builtin_prefetch(arow0 + k32 * 32 + 32, 0, 1);
            __builtin_prefetch(arow1 + k32 * 32 + 32, 0, 1);
        }

        // ---- 4 B fragments (contiguous), each feeding 2 wmma
        #pragma unroll
        for (int t = 0; t < 4; ++t) {
            const v16bf bf =
                *(const v16bf*)(bbase + ((size_t)(t * KT + k32) * 32 + lane) * 16);
            acc[0][t] = __builtin_amdgcn_wmma_f32_16x16x32_bf16(
                            false, af0, false, bf, (short)0, acc[0][t], false, false);
            acc[1][t] = __builtin_amdgcn_wmma_f32_16x16x32_bf16(
                            false, af1, false, bf, (short)0, acc[1][t], false, false);
        }
    }

    #pragma unroll
    for (int mi = 0; mi < 2; ++mi) {
        #pragma unroll
        for (int t = 0; t < 4; ++t) {
            const int col = strip_n * 64 + t * 16 + n;          // natural column
            const int w   = (t & 1) * 16 + n;                   // within 32-chunk
            const int colp = strip_n * 64 + (t >> 1) * 32 + permcol(w);
            const float bv = bias[col];
            #pragma unroll
            for (int r = 0; r < 8; ++r) {
                float v = acc[mi][t][r] + bv;
                if (RELU) v = fmaxf(v, 0.0f);
                const int row = (pair_m * 2 + mi) * 16 + r + hi * 8;
                if (WF32)  Cf [(size_t)row * N + col] = v;
                if (WBF16) Cbf[(size_t)row * N + (PERMBF ? colp : col)] = (__bf16)v;
            }
        }
    }
}

// ---------------------------------------------------------------------------
// Kernel 5: fused attention (flash-style online softmax).
// One wave per (batch, head, 16-query tile); 32 keys per iteration:
//   - next V tile staged global->LDS with global_load_async_to_lds_b128
//     (double buffer, s_wait_asynccnt), overlapping S-wmma + softmax;
//   - 4 wmma for S = (Q*scale)*K^T (Q from A-permuted storage: one v16bf),
//     pad mask, online row-max via __shfl_xor (width 16); lazy row-sum;
//   - P staged through LDS in A-permuted column order -> single contiguous
//     v16bf read (s_wait_dscnt); 4 wmma for O += P*V from LDS;
//   - output written A-permuted (consumed as A by the O-projection GEMM).
// ---------------------------------------------------------------------------
__global__ __launch_bounds__(32, 1) void
k_attn(const __bf16* __restrict__ Q,
       const __bf16* __restrict__ K,
       const __bf16* __restrict__ Vsw,
       const int*    __restrict__ tokens,
       __bf16* __restrict__ OUT)
{
    __shared__ __bf16 vstage[2][4 * 32 * 16];   // 2 x 4KB V double buffer
    __shared__ __bf16 pbuf[16 * 32];            // one 16x32 P tile (1 KB)

    const int lane = threadIdx.x;
    const int idx  = blockIdx.x;
    const int qt   = idx & 127;            // query tile (SEQ/16 = 128)
    const int h    = (idx >> 7) & 7;
    const int b    = idx >> 10;

    const int n    = lane & 15;
    const int hi   = lane >> 4;

    constexpr int KT = SEQ / 32;           // 64 key-steps

    // per-lane LDS byte offsets of the two V staging buffers
    const unsigned vlds[2] = {
        lds_off_u32(&vstage[0][0]) + (unsigned)lane * 32u,
        lds_off_u32(&vstage[1][0]) + (unsigned)lane * 32u
    };
    // per-lane global base of V fragment data for key-step k32
    const __bf16* vsrc_base =
        Vsw + ((size_t)((b * (SEQ / 32)) * NHEADS + h) * 4) * 512 + lane * 16;
    // stride between consecutive k32 tiles: NHEADS*4*512 elements
    const size_t vstride = (size_t)NHEADS * 4 * 512;

    // ---- Q A-fragments (permuted storage: one contiguous v16bf per chunk),
    //      pre-scaled by 1/sqrt(DK) (folds softmax scaling)
    v16bf aq[2];
    {
        const int qrow = b * SEQ + qt * 16 + n;       // A-layout row = lane&15
        const __bf16* qb = Q + (size_t)qrow * DMODEL + h * DK;
        #pragma unroll
        for (int c = 0; c < 2; ++c) {
            const v16bf q16 = *(const v16bf*)(qb + c * 32 + hi * 16);
            #pragma unroll
            for (int i = 0; i < 16; ++i)
                aq[c][i] = (__bf16)((float)q16[i] * 0.125f);
        }
    }

    // ---- prologue: stage V tile for k32 = 0
    #pragma unroll
    for (int t4 = 0; t4 < 4; ++t4)
        async_load_lds_b128(vlds[0] + t4 * 1024u, vsrc_base + t4 * 512);

    v8f o[4] = {};
    float mrow[8], lsum[8];
    #pragma unroll
    for (int r = 0; r < 8; ++r) { mrow[r] = -1e30f; lsum[r] = 0.0f; }

    // P staging: writer lane puts column K=n at perm position p0, K=16+n at p0+8
    const int p0 = (n & 7) | ((n & 8) << 1);

    for (int k32 = 0; k32 < KT; ++k32) {
        const int k0  = k32 * 32;
        const int cur = k32 & 1;

        // ---- kick off async stage of the NEXT V tile into the other buffer
        if (k32 + 1 < KT) {
            const __bf16* nsrc = vsrc_base + (size_t)(k32 + 1) * vstride;
            #pragma unroll
            for (int t4 = 0; t4 < 4; ++t4)
                async_load_lds_b128(vlds[cur ^ 1] + t4 * 1024u, nsrc + t4 * 512);
        }

        // ---------------- S = Q * K^T for two 16-key tiles ----------------
        v8f s0 = {}, s1 = {};
        #pragma unroll
        for (int c = 0; c < 2; ++c) {
            // K is natural order: two 8-value pieces (d = 8hi.., 16+8hi..), 16B aligned
            const __bf16* kr0 = K + (size_t)(b * SEQ + k0 + n) * DMODEL + h * DK + c * 32 + hi * 8;
            const __bf16* kr1 = K + (size_t)(b * SEQ + k0 + 16 + n) * DMODEL + h * DK + c * 32 + hi * 8;
            const v16bf bk0 = __builtin_shufflevector(*(const v8bf*)kr0, *(const v8bf*)(kr0 + 16),
                0, 1, 2, 3, 4, 5, 6, 7, 8, 9, 10, 11, 12, 13, 14, 15);
            const v16bf bk1 = __builtin_shufflevector(*(const v8bf*)kr1, *(const v8bf*)(kr1 + 16),
                0, 1, 2, 3, 4, 5, 6, 7, 8, 9, 10, 11, 12, 13, 14, 15);
            s0 = __builtin_amdgcn_wmma_f32_16x16x32_bf16(
                     false, aq[c], false, bk0, (short)0, s0, false, false);
            s1 = __builtin_amdgcn_wmma_f32_16x16x32_bf16(
                     false, aq[c], false, bk1, (short)0, s1, false, false);
        }

        // ---------------- key-pad mask (scale already folded into Q) --------
        const int pad0 = (tokens[b * SEQ + k0 + n]      == 0);
        const int pad1 = (tokens[b * SEQ + k0 + 16 + n] == 0);
        #pragma unroll
        for (int r = 0; r < 8; ++r) {
            if (pad0) s0[r] = -1e30f;
            if (pad1) s1[r] = -1e30f;
        }

        // ---------------- online softmax (row max only; lazy sum) -----------
        #pragma unroll
        for (int r = 0; r < 8; ++r) {
            float rmax = fmaxf(s0[r], s1[r]);
            #pragma unroll
            for (int off = 1; off < 16; off <<= 1)
                rmax = fmaxf(rmax, __shfl_xor(rmax, off, 16));
            const float mnew = fmaxf(mrow[r], rmax);
            const float cf   = __expf(mrow[r] - mnew);
            const float pp0  = __expf(s0[r] - mnew);
            const float pp1  = __expf(s1[r] - mnew);
            lsum[r] = lsum[r] * cf + pp0 + pp1;   // per-lane partial sum
            mrow[r] = mnew;
            o[0][r] *= cf; o[1][r] *= cf; o[2][r] *= cf; o[3][r] *= cf;
            s0[r] = pp0; s1[r] = pp1;
        }

        // ---------- P: C-layout -> A-permuted layout through LDS ------------
        #pragma unroll
        for (int r = 0; r < 8; ++r) {
            pbuf[(r + hi * 8) * 32 + p0]     = (__bf16)s0[r];
            pbuf[(r + hi * 8) * 32 + p0 + 8] = (__bf16)s1[r];
        }
        asm volatile("s_wait_dscnt 0" ::: "memory");   // wave-synchronous LDS RAW

        const v16bf pa = *(const v16bf*)&pbuf[n * 32 + hi * 16];

        // ---- wait for current V tile (next batch may still be in flight:
        //      async loads complete in order, so <=4 means ours are done)
        if (k32 + 1 < KT) asm volatile("s_wait_asynccnt 0x4" ::: "memory");
        else              asm volatile("s_wait_asynccnt 0x0" ::: "memory");

        // ---------------- O += P * V from LDS (contiguous v16bf) -------------
        #pragma unroll
        for (int t4 = 0; t4 < 4; ++t4) {
            const v16bf bv = *(const v16bf*)&vstage[cur][t4 * 512 + lane * 16];
            o[t4] = __builtin_amdgcn_wmma_f32_16x16x32_bf16(
                        false, pa, false, bv, (short)0, o[t4], false, false);
        }
    }

    // ---- final cross-lane row-sum reduction (once, not per iteration)
    #pragma unroll
    for (int r = 0; r < 8; ++r) {
        #pragma unroll
        for (int off = 1; off < 16; off <<= 1)
            lsum[r] += __shfl_xor(lsum[r], off, 16);
    }

    // ---------------- normalize + store bf16 (A-permuted columns) ------------
    #pragma unroll
    for (int r = 0; r < 8; ++r) {
        const float inv = 1.0f / lsum[r];
        const int row = b * SEQ + qt * 16 + r + hi * 8;
        #pragma unroll
        for (int t4 = 0; t4 < 4; ++t4) {
            const int w    = (t4 & 1) * 16 + n;
            const int colp = h * DK + (t4 >> 1) * 32 + permcol(w);
            OUT[(size_t)row * DMODEL + colp] = (__bf16)(o[t4][r] * inv);
        }
    }
}

// ---------------------------------------------------------------------------
// Kernel 6: y = LayerNorm(a + resid) * g + b ; f32 natural + optional bf16
// A-permuted output.
// ---------------------------------------------------------------------------
template<bool WF32, bool WBF16>
__global__ __launch_bounds__(256) void
k_add_ln(const float* __restrict__ a,
         const float* __restrict__ resid,
         const float* __restrict__ g,
         const float* __restrict__ bb,
         float* __restrict__ outf,
         __bf16* __restrict__ outbf)
{
    const int row  = blockIdx.x * blockDim.y + threadIdx.y;
    const int lane = threadIdx.x;
    const float* pa = a     + (size_t)row * DMODEL;
    const float* pr = resid + (size_t)row * DMODEL;

    float vals[16];
    float s = 0.0f, s2 = 0.0f;
    #pragma unroll
    for (int i = 0; i < 16; ++i) {
        const int c = lane + i * 32;
        const float v = pa[c] + pr[c];
        vals[i] = v; s += v; s2 += v * v;
    }
    #pragma unroll
    for (int off = 1; off < 32; off <<= 1) {
        s  += __shfl_xor(s,  off, 32);
        s2 += __shfl_xor(s2, off, 32);
    }
    const float mean = s * (1.0f / DMODEL);
    const float var  = s2 * (1.0f / DMODEL) - mean * mean;
    const float rstd = rsqrtf(var + LN_EPS);

    const int lp = permcol(lane);          // within-chunk permuted position
    #pragma unroll
    for (int i = 0; i < 16; ++i) {
        const int c = lane + i * 32;
        const float y = (vals[i] - mean) * rstd * g[c] + bb[c];
        if (WF32)  outf [(size_t)row * DMODEL + c] = y;
        if (WBF16) outbf[(size_t)row * DMODEL + i * 32 + lp] = (__bf16)y;
    }
}

// ---------------------------------------------------------------------------
// Host-side launch
// ---------------------------------------------------------------------------
extern "C" void kernel_launch(void* const* d_in, const int* in_sizes, int n_in,
                              void* d_out, int out_size, void* d_ws, size_t ws_size,
                              hipStream_t stream)
{
    const int*   tokens = (const int*)  d_in[0];
    const float* emb    = (const float*)d_in[1];
    const float* wq     = (const float*)d_in[2];
    const float* bq     = (const float*)d_in[3];
    const float* wk     = (const float*)d_in[4];
    const float* bk     = (const float*)d_in[5];
    const float* wv     = (const float*)d_in[6];
    const float* bv     = (const float*)d_in[7];
    const float* wo     = (const float*)d_in[8];
    const float* bo     = (const float*)d_in[9];
    const float* w1     = (const float*)d_in[10];
    const float* b1     = (const float*)d_in[11];
    const float* w2     = (const float*)d_in[12];
    const float* b2     = (const float*)d_in[13];
    const float* ln1_g  = (const float*)d_in[14];
    const float* ln1_b  = (const float*)d_in[15];
    const float* ln2_g  = (const float*)d_in[16];
    const float* ln2_b  = (const float*)d_in[17];

    // ---- workspace carve-up
    char* ws = (char*)d_ws;
    size_t off = 0;
    auto alloc = [&](size_t bytes) -> void* {
        void* p = ws + off;
        off += (bytes + 255) & ~(size_t)255;
        return p;
    };
    float*  xf     = (float*) alloc((size_t)NTOK * DMODEL * 4);
    __bf16* xbf    = (__bf16*)alloc((size_t)NTOK * DMODEL * 2);   // A-permuted
    __bf16* qbf    = (__bf16*)alloc((size_t)NTOK * DMODEL * 2);   // A-permuted
    __bf16* kbf    = (__bf16*)alloc((size_t)NTOK * DMODEL * 2);   // natural
    __bf16* vbf    = (__bf16*)alloc((size_t)NTOK * DMODEL * 2);   // natural
    __bf16* vsw    = (__bf16*)alloc((size_t)NTOK * DMODEL * 2);   // B-swizzled
    __bf16* avbf   = (__bf16*)alloc((size_t)NTOK * DMODEL * 2);   // A-permuted
    float*  attnf  = (float*) alloc((size_t)NTOK * DMODEL * 4);
    float*  hf     = (float*) alloc((size_t)NTOK * DMODEL * 4);
    __bf16* hbf    = (__bf16*)alloc((size_t)NTOK * DMODEL * 2);   // A-permuted
    __bf16* ff1bf  = (__bf16*)alloc((size_t)NTOK * DFF * 2);      // A-permuted
    float*  ff2f   = (float*) alloc((size_t)NTOK * DMODEL * 4);
    __bf16* wqb    = (__bf16*)alloc((size_t)DMODEL * DMODEL * 2);
    __bf16* wkb    = (__bf16*)alloc((size_t)DMODEL * DMODEL * 2);
    __bf16* wvb    = (__bf16*)alloc((size_t)DMODEL * DMODEL * 2);
    __bf16* wob    = (__bf16*)alloc((size_t)DMODEL * DMODEL * 2);
    __bf16* w1b    = (__bf16*)alloc((size_t)DMODEL * DFF * 2);
    __bf16* w2b    = (__bf16*)alloc((size_t)DFF * DMODEL * 2);
    (void)ws_size; (void)n_in; (void)in_sizes;

    float* outp  = (float*)d_out;
    float* maskp = (out_size >= NTOK * DMODEL + NTOK)
                 ? outp + (size_t)NTOK * DMODEL : nullptr;

    // ---- weight conversion + B-fragment swizzle (bf16)
    {
        const int t512 = (DMODEL / 16) * (DMODEL / 32) * 32;   // 16384
        const int tff1 = (DFF / 16) * (DMODEL / 32) * 32;      // 32768
        const int tff2 = (DMODEL / 16) * (DFF / 32) * 32;      // 32768
        k_swizzle_w<DMODEL, DMODEL><<<dim3((t512 + 255) / 256), dim3(256), 0, stream>>>(wq, wqb);
        k_swizzle_w<DMODEL, DMODEL><<<dim3((t512 + 255) / 256), dim3(256), 0, stream>>>(wk, wkb);
        k_swizzle_w<DMODEL, DMODEL><<<dim3((t512 + 255) / 256), dim3(256), 0, stream>>>(wv, wvb);
        k_swizzle_w<DMODEL, DMODEL><<<dim3((t512 + 255) / 256), dim3(256), 0, stream>>>(wo, wob);
        k_swizzle_w<DFF, DMODEL>   <<<dim3((tff1 + 255) / 256), dim3(256), 0, stream>>>(w1, w1b);
        k_swizzle_w<DMODEL, DFF>   <<<dim3((tff2 + 255) / 256), dim3(256), 0, stream>>>(w2, w2b);
    }

    // ---- embedding + PE + mask
    k_prep<<<dim3(NTOK), dim3(256), 0, stream>>>(tokens, emb, xf, xbf, maskp);

    // ---- Q/K/V projections (bf16 outputs; Q permuted for A-consumption)
    {
        dim3 blk(32, 4);
        dim3 grd(DMODEL / 64, NTOK / 128);    // each wave: 32 rows x 64 cols
        k_gemm<DMODEL, DMODEL, false, false, true, true ><<<grd, blk, 0, stream>>>(
            xbf, wqb, bq, (float*)nullptr, qbf);
        k_gemm<DMODEL, DMODEL, false, false, true, false><<<grd, blk, 0, stream>>>(
            xbf, wkb, bk, (float*)nullptr, kbf);
        k_gemm<DMODEL, DMODEL, false, false, true, false><<<grd, blk, 0, stream>>>(
            xbf, wvb, bv, (float*)nullptr, vbf);
    }

    // ---- V into attention B-fragment layout
    k_swizzle_v<<<dim3((NTOK * DMODEL / 16) / 256), dim3(256), 0, stream>>>(vbf, vsw);

    // ---- fused attention (writes avbf A-permuted)
    k_attn<<<dim3(BATCH * NHEADS * (SEQ / 16)), dim3(32), 0, stream>>>(
        qbf, kbf, vsw, tokens, avbf);

    // ---- output projection (f32)
    k_gemm<DMODEL, DMODEL, false, true, false, false><<<dim3(DMODEL / 64, NTOK / 128), dim3(32, 4), 0, stream>>>(
        avbf, wob, bo, attnf, (__bf16*)nullptr);

    // ---- LN1(attn + x) -> h (f32 natural + bf16 permuted)
    k_add_ln<true, true><<<dim3(NTOK / 8), dim3(32, 8), 0, stream>>>(
        attnf, xf, ln1_g, ln1_b, hf, hbf);

    // ---- FFN: relu(h@w1+b1) -> bf16 permuted ; @w2+b2 -> f32
    k_gemm<DFF, DMODEL, true, false, true, true><<<dim3(DFF / 64, NTOK / 128), dim3(32, 4), 0, stream>>>(
        hbf, w1b, b1, (float*)nullptr, ff1bf);
    k_gemm<DMODEL, DFF, false, true, false, false><<<dim3(DMODEL / 64, NTOK / 128), dim3(32, 4), 0, stream>>>(
        ff1bf, w2b, b2, ff2f, (__bf16*)nullptr);

    // ---- LN2(h + ff) -> d_out (f32)
    k_add_ln<true, false><<<dim3(NTOK / 8), dim3(32, 8), 0, stream>>>(
        ff2f, hf, ln2_g, ln2_b, outp, (__bf16*)nullptr);
}
